// QuantumConv2d_7868380086683
// MI455X (gfx1250) — compile-verified, matrix operand order checked by reading, not run on hardware
//
#include <hip/hip_runtime.h>
#include <stdint.h>

// ---------------------------------------------------------------------------
// QuantumConv2d for MI455X (gfx1250, wave32, WMMA).
//   k=3, q=9, dim=512, N = 25600 patches.
//   phase[p,n] = S_n + (S_n^2 - Q_p)/2 ;  psi = exp(-0.5i*phase)/sqrt(512)
//   psi_out = psi @ U^T : complex 512x512 GEMM via 3-mult (Karatsuba) f16
//   WMMA with f32 accumulation:
//       P1 = Ar*Br, P2 = Ai*Bi, P3 = (Ar+Ai)*(Br+Bi)
//       Re = P1 - P2,  Im = P3 - P1 - P2
//   out[p,j] = sum_m |psi_out[p,m]|^2 * bit_{8-j}(m)
// U[m,n] = (-i)^popcount(rho(m)^n) * prod_i (c_i | s_i); built per launch
// into d_ws as three f16 planes (Re, Im, Re+Im), each pre-swizzled into the
// exact WMMA 16-bit B-fragment order so inner-loop loads are pure b128 with
// zero unpack ALU. 1.5 MB total -> resident in the 192 MB L2.
// ---------------------------------------------------------------------------

typedef __attribute__((ext_vector_type(16))) _Float16 v16h;
typedef __attribute__((ext_vector_type(8)))  float    v8f;

union F16FRAG { uint4 q[2]; v16h h; };   // 16 halves = 32 B = 2 x b128

__device__ __forceinline__ unsigned short f2h(float x) {
  _Float16 h = (_Float16)x;
  return __builtin_bit_cast(unsigned short, h);
}

// fragment index mapping for 16-bit operands (ISA 7.12.2):
//   K = (j<8 ? j : j+8) + 8*part  <=>  part=(K>>3)&1, j=(K&7)+((K>>4)<<3)

// ---------------------------------------------------------------------------
// Setup: build B = U^T as three half planes in fragment order.
//   fi = [ct = m>>4][kc = n>>5][part][ncol = m&15][j]
// ---------------------------------------------------------------------------
__global__ __launch_bounds__(256) void build_u_kernel(const float* __restrict__ wgt,
                                                      uint16_t* __restrict__ Bre,
                                                      uint16_t* __restrict__ Bim,
                                                      uint16_t* __restrict__ Bsm) {
  int idx = blockIdx.x * 256 + threadIdx.x;
  if (idx >= 512 * 512) return;
  int m = idx >> 9;
  int n = idx & 511;

  float cc[9], ss[9];
#pragma unroll
  for (int i = 0; i < 9; ++i) __sincosf(0.5f * wgt[i], &ss[i], &cc[i]);

  // ring row permutation: C(8,0) first, then C(7,8) ... C(0,1)
  int mp = m ^ ((m >> 8) & 1);
#pragma unroll
  for (int i = 7; i >= 0; --i) mp ^= ((mp >> i) & 1) << (i + 1);

  int xo = mp ^ n;
  float mag = 1.f;
#pragma unroll
  for (int i = 0; i < 9; ++i) mag *= ((xo >> i) & 1) ? ss[i] : cc[i];

  int d = __popc(xo) & 3;                 // (-i)^d
  float re = 0.f, im = 0.f;
  if      (d == 0) re =  mag;
  else if (d == 1) im = -mag;
  else if (d == 2) re = -mag;
  else             im =  mag;

  int K = n & 31, kc = n >> 5;
  int prt = (K >> 3) & 1;
  int j   = (K & 7) + ((K >> 4) << 3);
  int fi = ((((m >> 4) * 16 + kc) * 2 + prt) * 16 + (m & 15)) * 16 + j;
  Bre[fi] = f2h(re);
  Bim[fi] = f2h(im);
  Bsm[fi] = f2h(re + im);
}

// ---------------------------------------------------------------------------
// Main kernel: one workgroup (8 waves, 256 threads) per 16 patches.
// ---------------------------------------------------------------------------
__global__ __launch_bounds__(256) void qconv_kernel(const float* __restrict__ x,
                                                    const uint16_t* __restrict__ Bre,
                                                    const uint16_t* __restrict__ Bim,
                                                    const uint16_t* __restrict__ Bsm,
                                                    float* __restrict__ out) {
  // planes: psi fragments [kc(16)][part(2)][row(16)][j(16)], 16 KB each.
  // After the GEMM the same memory is reused for probs + partial sums.
  __shared__ union SMem {
    struct { uint16_t re[8192], im[8192], sm[8192]; } pl;   // 48 KB
    struct { float probs[8192]; float part2[2304]; } rd;    // 41 KB
  } smem;
  __shared__ float pv[16][9];

  const int tid  = threadIdx.x;
  const int lane = tid & 31;
  const int wv   = tid >> 5;       // wave 0..7 owns columns [wv*64, wv*64+64)
  const int part = lane >> 4;
  const int row  = lane & 15;

  // ---- load the 16 patches of this block ----
  if (tid < 144) {
    int p = tid / 9, i = tid - p * 9;
    int P = blockIdx.x * 16 + p;
    int b = P / 100, rem = P - b * 100;
    int gy = rem / 10, gx = rem - gy * 10;
    int r = i / 3, c = i - r * 3;
    pv[p][i] = x[b * 900 + (gy * 3 + r) * 30 + (gx * 3 + c)];
  }
  __syncthreads();

  // ---- psi = exp(-0.5i*phase)/sqrt(512), phase = S + (S^2 - Q)/2 ----
  const float INV = 0.04419417382415922f;   // 1/sqrt(512)
  for (int idx = tid; idx < 8192; idx += 256) {
    int p = idx >> 9, n = idx & 511;
    float S = 0.f, Q = 0.f;
#pragma unroll
    for (int i = 0; i < 9; ++i) {
      float v = pv[p][i];
      S += ((n >> i) & 1) ? v : -v;
      Q += v * v;
    }
    float ph = S + 0.5f * (S * S - Q);
    float sn, cn;
    __sincosf(0.5f * ph, &sn, &cn);
    float vre = INV * cn;
    float vim = -INV * sn;
    int kc = n >> 5, K = n & 31;
    int prt = (K >> 3) & 1;
    int j   = (K & 7) + ((K >> 4) << 3);
    int base = ((kc * 2 + prt) * 16 + p) * 16 + j;
    smem.pl.re[base] = f2h(vre);
    smem.pl.im[base] = f2h(vim);
    smem.pl.sm[base] = f2h(vre + vim);
  }
  __syncthreads();

  // ---- 3-mult complex GEMM: 16x512 (psi) x 512x512 (U^T) ----
  v8f P1[4] = {};
  v8f P2[4] = {};
  v8f P3[4] = {};

  for (int kc = 0; kc < 16; ++kc) {
    int aoff = ((kc * 2 + part) * 16 + row) * 16;
    F16FRAG ar, ai, as;
    { const uint4* p_ = (const uint4*)(smem.pl.re + aoff); ar.q[0] = p_[0]; ar.q[1] = p_[1]; }
    { const uint4* p_ = (const uint4*)(smem.pl.im + aoff); ai.q[0] = p_[0]; ai.q[1] = p_[1]; }
    { const uint4* p_ = (const uint4*)(smem.pl.sm + aoff); as.q[0] = p_[0]; as.q[1] = p_[1]; }

    if (kc + 1 < 16) {   // pull next K chunk of U toward the WGP
      int nfi = ((((wv << 2) * 16 + (kc + 1)) * 2 + part) * 16 + row) * 16;
      __builtin_prefetch(Bre + nfi, 0, 1);
      __builtin_prefetch(Bim + nfi, 0, 1);
      __builtin_prefetch(Bsm + nfi, 0, 1);
    }

#pragma unroll
    for (int ct = 0; ct < 4; ++ct) {
      int fi = (((((wv << 2) + ct) * 16 + kc) * 2 + part) * 16 + row) * 16;
      F16FRAG br, bi, bs;
      { const uint4* p_ = (const uint4*)(Bre + fi); br.q[0] = p_[0]; br.q[1] = p_[1]; }
      { const uint4* p_ = (const uint4*)(Bim + fi); bi.q[0] = p_[0]; bi.q[1] = p_[1]; }
      { const uint4* p_ = (const uint4*)(Bsm + fi); bs.q[0] = p_[0]; bs.q[1] = p_[1]; }

      P1[ct] = __builtin_amdgcn_wmma_f32_16x16x32_f16(
          false, ar.h, false, br.h, (short)0, P1[ct], false, false);
      P2[ct] = __builtin_amdgcn_wmma_f32_16x16x32_f16(
          false, ai.h, false, bi.h, (short)0, P2[ct], false, false);
      P3[ct] = __builtin_amdgcn_wmma_f32_16x16x32_f16(
          false, as.h, false, bs.h, (short)0, P3[ct], false, false);
    }
  }

  // ---- probs = |psi_out|^2 into LDS (reuse planes as float[16][512]) ----
  __syncthreads();                       // all psi reads done before overwrite
#pragma unroll
  for (int ct = 0; ct < 4; ++ct) {
    int m = (((wv << 2) + ct) << 4) + row;
#pragma unroll
    for (int r = 0; r < 8; ++r) {
      float p1 = P1[ct][r], p2 = P2[ct][r], p3 = P3[ct][r];
      float re = p1 - p2;
      float im = p3 - p1 - p2;
      int rr = r + (part << 3);          // C layout: M = r + 8*(lane>=16)
      smem.rd.probs[(rr << 9) + m] = re * re + im * im;
    }
  }
  __syncthreads();

  // ---- stage 1: per-(patch, m-group) partials using bit structure of m ----
  {
    int p = tid >> 4, mg = tid & 15;     // m = mg*32 + mm ; bits 8..5 from mg
    const float* pr = smem.rd.probs + (p << 9) + (mg << 5);
    float tot = 0.f, sb0 = 0.f, sb1 = 0.f, sb2 = 0.f, sb3 = 0.f, sb4 = 0.f;
#pragma unroll
    for (int mm = 0; mm < 32; ++mm) {
      float v = pr[mm];
      tot += v;
      if (mm & 1)  sb0 += v;
      if (mm & 2)  sb1 += v;
      if (mm & 4)  sb2 += v;
      if (mm & 8)  sb3 += v;
      if (mm & 16) sb4 += v;
    }
    float sb[5] = {sb0, sb1, sb2, sb3, sb4};
#pragma unroll
    for (int j = 0; j < 9; ++j) {
      // out bit position: shift = 8-j. j<4 -> mg bit (3-j); j>=4 -> mm bit (8-j)
      float pj = (j < 4) ? (((mg >> (3 - j)) & 1) ? tot : 0.f) : sb[8 - j];
      smem.rd.part2[(p * 9 + j) * 16 + mg] = pj;
    }
  }
  __syncthreads();

  // ---- stage 2: 16-way sums + scatter to output layout ----
  if (tid < 144) {
    int p = tid / 9, j = tid - p * 9;
    const float* q_ = smem.rd.part2 + (p * 9 + j) * 16;
    float s = 0.f;
#pragma unroll
    for (int mg = 0; mg < 16; ++mg) s += q_[mg];
    int P = blockIdx.x * 16 + p;
    int b = P / 100, rem = P - b * 100;
    int gy = rem / 10, gx = rem - gy * 10;
    int r = j / 3, c = j - r * 3;
    out[b * 900 + (gy * 3 + r) * 30 + (gx * 3 + c)] = s;
  }
}

// ---------------------------------------------------------------------------
extern "C" void kernel_launch(void* const* d_in, const int* in_sizes, int n_in,
                              void* d_out, int out_size, void* d_ws, size_t ws_size,
                              hipStream_t stream) {
  const float* x = (const float*)d_in[0];       // (256,30,30) f32
  const float* w = (const float*)d_in[1];       // (3,3) f32
  uint16_t* base = (uint16_t*)d_ws;             // 3 x 512 KB = 1.5 MB
  uint16_t* Bre = base;
  uint16_t* Bim = base + 262144;
  uint16_t* Bsm = base + 524288;

  build_u_kernel<<<1024, 256, 0, stream>>>(w, Bre, Bim, Bsm);
  qconv_kernel<<<1600, 256, 0, stream>>>(x, Bre, Bim, Bsm, (float*)d_out);
}
